// SimpleLSTM_66675072303562
// MI455X (gfx1250) — matrix operand order repaired and checked by reading, not compiled, not run on hardware
//
#include <hip/hip_runtime.h>
#include <hip/hip_bf16.h>

// Problem dims (fixed by reference)
#define BB   64
#define DD   512
#define TT   512
#define HH   1024
#define G4   4096      // 4*H
#define KCAT 1536      // D + H

// LDS weight tile geometry (TDM-written, hardware-padded)
// unpadded row = 1536 bf16 = 3072 B; TDM inserts 16 B after every 1024 B
#define ROW_BYTES   3120u                  // 3072 + 3*16
#define CHUNK_BYTES 1040u                  // 1024 + 16 (padded 1024-B chunk)
#define TILE_BYTES  (16u * ROW_BYTES)      // one gate tile: 49920 B
#define LDS_BYTES   (4u * TILE_BYTES)      // 199680 B < 320 KB/WGP

typedef __attribute__((ext_vector_type(16))) __bf16 bf16x16;
typedef __attribute__((ext_vector_type(8)))  __bf16 bf16x8;
typedef __attribute__((ext_vector_type(8)))  float  f32x8;
typedef __attribute__((ext_vector_type(4)))  unsigned int u32x4;
typedef __attribute__((ext_vector_type(8)))  int i32x8;
typedef __attribute__((ext_vector_type(4)))  int i32x4;

// ---------------------------------------------------------------------------
// Fragment loads matching CDNA5 WMMA VGPR layouts (ISA 7.12.2)
// A (16x32 bf16): lane l row M=l&15; half=l>>4 selects K-suboffset.
// ---------------------------------------------------------------------------
__device__ __forceinline__ bf16x16 load_a_frag(const __bf16* row, int half) {
  bf16x8 lo = *(const bf16x8*)(row + half * 8);
  bf16x8 hi = *(const bf16x8*)(row + 16 + half * 8);
  bf16x16 r;
#pragma unroll
  for (int i = 0; i < 8; ++i) { r[i] = lo[i]; r[i + 8] = hi[i]; }
  return r;
}

// B (32x16 bf16) from the row-major LDS tile: lane l = col N (l&15),
// K = 8*(l>>4)+0..7 and 16+8*(l>>4)+0..7 of the weight row. Two ds_load_b128.
__device__ __forceinline__ bf16x16 load_b_frag_lds(const char* p) {
  bf16x8 lo = *(const bf16x8*)(p);
  bf16x8 hi = *(const bf16x8*)(p + 32);
  bf16x16 r;
#pragma unroll
  for (int i = 0; i < 8; ++i) { r[i] = lo[i]; r[i + 8] = hi[i]; }
  return r;
}

__device__ __forceinline__ f32x8 wmma_bf16(bf16x16 a, bf16x16 b, f32x8 c) {
  return __builtin_amdgcn_wmma_f32_16x16x32_bf16(
      false, a, false, b, (short)0, c, false, false);
}

__device__ __forceinline__ float sigm_(float x) { return 1.0f / (1.0f + __expf(-x)); }
__device__ __forceinline__ float tanh_(float x) {
  float e = __expf(-2.0f * x);
  return (1.0f - e) / (1.0f + e);
}

// ---------------------------------------------------------------------------
// Grid-wide barrier (all 64 blocks resident by construction).
// ---------------------------------------------------------------------------
__device__ __forceinline__ void grid_barrier(unsigned* bar, unsigned nblk, unsigned step) {
  __syncthreads();
  if (threadIdx.x == 0) {
    __threadfence();
    unsigned ticket = atomicAdd(&bar[0], 1u);
    if (ticket == nblk - 1u) {
      bar[0] = 0u;
      __threadfence();
      atomicAdd(&bar[1], 1u);
    } else {
      while (atomicAdd(&bar[1], 0u) <= step) {
        __builtin_amdgcn_s_sleep(2);
      }
    }
    __threadfence();
  }
  __syncthreads();
}

// ---------------------------------------------------------------------------
// Prep kernels
// ---------------------------------------------------------------------------
__global__ void lstm_prep_x(const float* __restrict__ x, __bf16* __restrict__ xbf) {
  size_t idx = (size_t)blockIdx.x * blockDim.x + threadIdx.x;
  const size_t N = (size_t)TT * BB * DD;
  if (idx >= N) return;
  int d = (int)(idx % DD);
  int b = (int)((idx / DD) % BB);
  int t = (int)(idx / ((size_t)DD * BB));
  xbf[idx] = (__bf16)x[((size_t)b * DD + d) * TT + t];   // x is (B, D, T)
}

__global__ void lstm_prep_w(const float* __restrict__ Wih, const float* __restrict__ Whh,
                            __bf16* __restrict__ Wcat) {
  size_t idx = (size_t)blockIdx.x * blockDim.x + threadIdx.x;
  const size_t N = (size_t)G4 * KCAT;
  if (idx >= N) return;
  int k = (int)(idx % KCAT);
  int g = (int)(idx / KCAT);
  float v = (k < DD) ? Wih[(size_t)g * DD + k] : Whh[(size_t)g * HH + (k - DD)];
  Wcat[idx] = (__bf16)v;
}

__global__ void lstm_init(__bf16* __restrict__ hs, unsigned* __restrict__ bar) {
  int idx = blockIdx.x * blockDim.x + threadIdx.x;
  if (idx < BB * HH) hs[idx] = (__bf16)0.0f;             // h_0 = 0 (slice t=0)
  if (idx < 8) bar[idx] = 0u;
}

// ---------------------------------------------------------------------------
// Persistent LSTM scan kernel: 64 blocks (one per h-tile) x 128 threads
// (4 waves = 4 batch-tiles). Cell state c stays in registers for all 512
// steps. Each wave TDM-DMAs its gate's 16x1536 bf16 weight tile into LDS
// once (195 KB total, fits only in CDNA5's 320 KB/WGP), with hardware
// padding (16 B per 1024 B) so steady-state ds_load_b128 reads are
// bank-conflict-free and use constant DS offsets.
// ---------------------------------------------------------------------------
__global__ void __launch_bounds__(128, 1)
lstm_scan(const __bf16* __restrict__ xbf, const __bf16* __restrict__ Wcat,
          const float* __restrict__ b_ih, const float* __restrict__ b_hh,
          __bf16* __restrict__ hs, unsigned* __restrict__ bar) {
  extern __shared__ char smem[];        // 4 gate tiles, TDM-padded rows

  const int lane  = threadIdx.x & 31;
  const int wave  = threadIdx.x >> 5;   // 0..3  -> batch tile (and fill-gate)
  const int htile = blockIdx.x;         // 0..63 -> hidden tile
  const int half  = lane >> 4;
  const int col   = lane & 15;
  const int hh    = htile * 16 + col;   // hidden column this lane owns (C/D)
  const int bA    = wave * 16 + col;    // batch row this lane supplies (A)

  // ---- TDM: async-DMA gate `wave`'s weight tile (16 rows x 1536 bf16) ----
  // No static LDS in this kernel => dynamic segment starts at LDS offset 0.
  {
    unsigned long long ga =
        (unsigned long long)(const void*)(Wcat + (size_t)(wave * HH + htile * 16) * KCAT);
    u32x4 g0;
    g0[0] = 1u;                                      // count=1 (valid user D#)
    g0[1] = (unsigned)wave * TILE_BYTES;             // lds_addr
    g0[2] = (unsigned)(ga & 0xFFFFFFFFull);          // global_addr[31:0]
    g0[3] = (unsigned)((ga >> 32) & 0x1FFFFFFull) | (2u << 30);  // addr[56:32] | type=2
    i32x8 g1;
    g1[0] = (1 << 16)        // data_size = 2 bytes
          | (1 << 20)        // pad_enable
          | (7 << 22)        // pad_interval: 256 dwords = 1024 B
          | (3 << 25);       // pad_amount: 4 dwords = 16 B
    g1[1] = (KCAT & 0xFFFF) << 16;                   // tensor_dim0[15:0]
    g1[2] = (16 << 16);                              // tensor_dim1[15:0]=16, dim0[31:16]=0
    g1[3] = (KCAT & 0xFFFF) << 16;                   // tile_dim0 = 1536
    g1[4] = 16;                                      // tile_dim1 = 16, tile_dim2 = 0 (2-D)
    g1[5] = KCAT;                                    // tensor_dim0_stride[31:0]
    g1[6] = (KCAT & 0xFFFF) << 16;                   // tensor_dim1_stride[15:0]=1536
    g1[7] = 0;                                       // tensor_dim1_stride[47:16]
    i32x4 g2 = {0, 0, 0, 0};
    i32x4 g3 = {0, 0, 0, 0};
    i32x8 g4 = {0, 0, 0, 0, 0, 0, 0, 0};             // unused trailing group (clang-23 form)
    __builtin_amdgcn_tensor_load_to_lds(g0, g1, g2, g3, g4, 0);
  }
  __builtin_amdgcn_s_wait_tensorcnt(0);
  __syncthreads();

  // Per-gate, per-lane B-fragment base pointers inside the padded LDS tiles.
  // Lane reads row (gate*16 + col), k-suboffset half*8 elements = half*16 B.
  const char* b0 = smem + (size_t)(0u * TILE_BYTES) + (unsigned)col * ROW_BYTES + half * 16;
  const char* b1 = smem + (size_t)(1u * TILE_BYTES) + (unsigned)col * ROW_BYTES + half * 16;
  const char* b2 = smem + (size_t)(2u * TILE_BYTES) + (unsigned)col * ROW_BYTES + half * 16;
  const char* b3 = smem + (size_t)(3u * TILE_BYTES) + (unsigned)col * ROW_BYTES + half * 16;

  const float bias_i = b_ih[0 * HH + hh] + b_hh[0 * HH + hh];
  const float bias_f = b_ih[1 * HH + hh] + b_hh[1 * HH + hh];
  const float bias_g = b_ih[2 * HH + hh] + b_hh[2 * HH + hh];
  const float bias_o = b_ih[3 * HH + hh] + b_hh[3 * HH + hh];

  float creg[8];
#pragma unroll
  for (int v = 0; v < 8; ++v) creg[v] = 0.0f;

  for (int t = 0; t < TT; ++t) {
    f32x8 ai, af, ag, ao;
#pragma unroll
    for (int v = 0; v < 8; ++v) { ai[v] = bias_i; af[v] = bias_f; ag[v] = bias_g; ao[v] = bias_o; }

    // K = 0..511 : input projection. k-tiles 0..15 live in padded chunk 0.
    const __bf16* xrow = xbf + ((size_t)t * BB + bA) * DD;
#pragma unroll 4
    for (int j = 0; j < 16; ++j) {
      bf16x16 a = load_a_frag(xrow + j * 32, half);
      ai = wmma_bf16(a, load_b_frag_lds(b0 + j * 64), ai);
      af = wmma_bf16(a, load_b_frag_lds(b1 + j * 64), af);
      ag = wmma_bf16(a, load_b_frag_lds(b2 + j * 64), ag);
      ao = wmma_bf16(a, load_b_frag_lds(b3 + j * 64), ao);
    }

    // K = 512..1535 : recurrent projection. k-tiles 16..47 = padded chunks 1,2.
    const __bf16* hrow = hs + ((size_t)t * BB + bA) * HH;
#pragma unroll 4
    for (int j = 0; j < 16; ++j) {
      bf16x16 a = load_a_frag(hrow + j * 32, half);
      ai = wmma_bf16(a, load_b_frag_lds(b0 + CHUNK_BYTES + j * 64), ai);
      af = wmma_bf16(a, load_b_frag_lds(b1 + CHUNK_BYTES + j * 64), af);
      ag = wmma_bf16(a, load_b_frag_lds(b2 + CHUNK_BYTES + j * 64), ag);
      ao = wmma_bf16(a, load_b_frag_lds(b3 + CHUNK_BYTES + j * 64), ao);
    }
#pragma unroll 4
    for (int j = 0; j < 16; ++j) {
      bf16x16 a = load_a_frag(hrow + 512 + j * 32, half);
      ai = wmma_bf16(a, load_b_frag_lds(b0 + 2 * CHUNK_BYTES + j * 64), ai);
      af = wmma_bf16(a, load_b_frag_lds(b1 + 2 * CHUNK_BYTES + j * 64), af);
      ag = wmma_bf16(a, load_b_frag_lds(b2 + 2 * CHUNK_BYTES + j * 64), ag);
      ao = wmma_bf16(a, load_b_frag_lds(b3 + 2 * CHUNK_BYTES + j * 64), ao);
    }

    // Gate nonlinearity + state update. C/D layout: VGPR v -> M = v + 8*half.
    __bf16* hnext = hs + (size_t)(t + 1) * BB * HH;
#pragma unroll
    for (int v = 0; v < 8; ++v) {
      int brow = wave * 16 + v + 8 * half;
      float iv = sigm_(ai[v]);
      float fv = sigm_(af[v]);
      float gv = tanh_(ag[v]);
      float ov = sigm_(ao[v]);
      float ct = fv * creg[v] + iv * gv;
      creg[v] = ct;
      float hv = ov * tanh_(ct);
      hnext[(size_t)brow * HH + hh] = (__bf16)hv;
    }

    grid_barrier(bar, gridDim.x, (unsigned)t);
  }
}

// ---------------------------------------------------------------------------
// Output projection: one wave per (b, t); out[b*T + t] = hs[t+1][b] . W_out + b_out
// ---------------------------------------------------------------------------
__global__ void lstm_proj(const __bf16* __restrict__ hs, const float* __restrict__ W_out,
                          const float* __restrict__ b_out, float* __restrict__ out) {
  int gwave = (int)((blockIdx.x * blockDim.x + threadIdx.x) >> 5);
  int lane  = threadIdx.x & 31;
  if (gwave >= BB * TT) return;
  int b = gwave / TT;
  int t = gwave % TT;
  const __bf16* h = hs + ((size_t)(t + 1) * BB + b) * HH;
  float s = 0.0f;
#pragma unroll 4
  for (int j = lane; j < HH; j += 32) s += (float)h[j] * W_out[j];
#pragma unroll
  for (int off = 16; off > 0; off >>= 1) s += __shfl_xor(s, off, 32);
  if (lane == 0) out[(size_t)b * TT + t] = s + b_out[0];
}

// ---------------------------------------------------------------------------
extern "C" void kernel_launch(void* const* d_in, const int* in_sizes, int n_in,
                              void* d_out, int out_size, void* d_ws, size_t ws_size,
                              hipStream_t stream) {
  const float* x     = (const float*)d_in[0];
  const float* W_ih  = (const float*)d_in[1];
  const float* W_hh  = (const float*)d_in[2];
  const float* b_ih  = (const float*)d_in[3];
  const float* b_hh  = (const float*)d_in[4];
  const float* W_out = (const float*)d_in[5];
  const float* b_out = (const float*)d_in[6];
  float* out = (float*)d_out;

  // Workspace carve-up (all 256-B aligned)
  char* ws = (char*)d_ws;
  unsigned* bar = (unsigned*)ws;                                   // 256 B
  __bf16* xbf  = (__bf16*)(ws + 256);                              // T*B*D bf16 = 32 MB
  __bf16* Wcat = (__bf16*)(ws + 256 + (size_t)TT * BB * DD * 2);   // 4H*KCAT bf16 = 12 MB
  __bf16* hs   = (__bf16*)(ws + 256 + (size_t)TT * BB * DD * 2
                                   + (size_t)G4 * KCAT * 2);       // (T+1)*B*H bf16 = 64.1 MB

  {
    size_t n = (size_t)TT * BB * DD;
    lstm_prep_x<<<(unsigned)((n + 255) / 256), 256, 0, stream>>>(x, xbf);
  }
  {
    size_t n = (size_t)G4 * KCAT;
    lstm_prep_w<<<(unsigned)((n + 255) / 256), 256, 0, stream>>>(W_ih, W_hh, Wcat);
  }
  lstm_init<<<(BB * HH + 255) / 256, 256, 0, stream>>>(hs, bar);

  // Persistent scan: 64 blocks x 128 threads, 195 KB dynamic LDS per block.
  lstm_scan<<<HH / 16, 128, LDS_BYTES, stream>>>(xbf, Wcat, b_ih, b_hh, hs, bar);

  // Projection: one wave per (b, t) output element
  lstm_proj<<<(BB * TT * 32 + 255) / 256, 256, 0, stream>>>(hs, W_out, b_out, out);
}
